// CRDLoss_32676111188335
// MI455X (gfx1250) — compile-verified, matrix operand order checked by reading, not run on hardware
//
#include <hip/hip_runtime.h>
#include <math.h>

typedef __attribute__((ext_vector_type(16))) _Float16 v16h;
typedef __attribute__((ext_vector_type(8)))  float    v8f;
typedef __attribute__((ext_vector_type(4)))  float    vf4;

#define BSZ      256
#define FEAT     128
#define KTOT     4097          // K+1
#define N_DATA_F 500000.0f
#define T_INV    (1.0f / 0.07f)

// ---------------------------------------------------------------------------
// Projection GEMM with WMMA:  out[M,128] = f16(A[M,Kdim]) @ f16(W[Kdim,128]) + b
// grid.x = M/16, block = 256 (8 waves); wave w owns the 16x16 tile at cols 16w.
// f32 accumulate via v_wmma_f32_16x16x32_f16.
// ---------------------------------------------------------------------------
__global__ __launch_bounds__(256) void proj_wmma(
    const float* __restrict__ A, const float* __restrict__ W,
    const float* __restrict__ bias, float* __restrict__ out, int Kdim)
{
    const int lane = threadIdx.x & 31;
    const int wave = threadIdx.x >> 5;
    const int m0 = blockIdx.x << 4;
    const int n0 = wave << 4;
    const int g  = lane >> 4;      // lane half selects K sub-block
    const int ml = lane & 15;      // row (A) / col (B)

    v8f c = {};
    const float* arow = A + (m0 + ml) * Kdim;
    for (int k0 = 0; k0 < Kdim; k0 += 32) {
        v16h a, b;
        // A 16x32 f16 fragment: elems 0..7 -> K = k0+8g .. +7 ; 8..15 -> K = k0+16+8g ..
        const float* ap = arow + k0 + 8 * g;
#pragma unroll
        for (int e = 0; e < 8; ++e) a[e]     = (_Float16)ap[e];
#pragma unroll
        for (int e = 0; e < 8; ++e) a[8 + e] = (_Float16)ap[16 + e];
        // B 32x16 f16 fragment (lane = N, same K pattern), W row-major [K][128]
        const float* wp = W + (k0 + 8 * g) * FEAT + (n0 + ml);
#pragma unroll
        for (int e = 0; e < 8; ++e) b[e]     = (_Float16)wp[e * FEAT];
#pragma unroll
        for (int e = 0; e < 8; ++e) b[8 + e] = (_Float16)wp[(16 + e) * FEAT];
        c = __builtin_amdgcn_wmma_f32_16x16x32_f16(
                false, a, false, b, (short)0, c, false, false);
    }
    // C/D layout: VGPR r -> row m0 + r + 8*g, col n0 + ml
    const int col = n0 + ml;
    const float bv = bias[col];
#pragma unroll
    for (int r = 0; r < 8; ++r)
        out[(m0 + r + 8 * g) * FEAT + col] = c[r] + bv;
}

// ---------------------------------------------------------------------------
// Row L2 normalize, one wave per 128-float row (in place)
// ---------------------------------------------------------------------------
__global__ __launch_bounds__(256) void l2norm_rows(float* __restrict__ x, int rows)
{
    const int row = (blockIdx.x << 3) + (threadIdx.x >> 5);
    if (row >= rows) return;
    const int lane = threadIdx.x & 31;
    float* p = x + (size_t)row * FEAT + lane * 4;
    vf4 v = *(vf4*)p;
    float s = v.x * v.x + v.y * v.y + v.z * v.z + v.w * v.w;
#pragma unroll
    for (int off = 16; off; off >>= 1) s += __shfl_xor(s, off, 32);
    const float inv = 1.0f / sqrtf(s);
    v.x *= inv; v.y *= inv; v.z *= inv; v.w *= inv;
    *(vf4*)p = v;
}

// ---------------------------------------------------------------------------
// NCE: out[b,k] = exp(dot(mem[cidx[b,k]], emb[b]) / T)
// block = one batch element (8 waves); wave handles one k per iteration:
// 32 lanes x float4 = one coalesced 512B row, cross-lane shfl reduction.
// ---------------------------------------------------------------------------
__global__ __launch_bounds__(256) void nce_kernel(
    const float* __restrict__ mem, const float* __restrict__ emb,
    const int* __restrict__ cidx, float* __restrict__ outx)
{
    const int b    = blockIdx.x;
    const int lane = threadIdx.x & 31;
    const int wave = threadIdx.x >> 5;
    const vf4 e = *(const vf4*)(emb + (size_t)b * FEAT + lane * 4);
    const int* ci = cidx + (size_t)b * KTOT;
    float* ob = outx + (size_t)b * KTOT;
    for (int k = wave; k < KTOT; k += 8) {
        const int row = ci[k];
        if (k + 8 < KTOT)   // pull next gathered row toward L2/L0
            __builtin_prefetch(mem + (size_t)ci[k + 8] * FEAT + lane * 4, 0, 3);
        const vf4 m = *(const vf4*)(mem + (size_t)row * FEAT + lane * 4);
        float d = m.x * e.x + m.y * e.y + m.z * e.z + m.w * e.w;
#pragma unroll
        for (int off = 16; off; off >>= 1) d += __shfl_xor(d, off, 32);
        if (lane == 0) ob[k] = __expf(d * T_INV);
    }
}

// ---------------------------------------------------------------------------
// Deterministic reductions (fixed grids, no float atomics)
// ---------------------------------------------------------------------------
__device__ __forceinline__ float block_reduce(float v)
{
    __shared__ float red[8];
#pragma unroll
    for (int off = 16; off; off >>= 1) v += __shfl_xor(v, off, 32);
    const int lane = threadIdx.x & 31, wave = threadIdx.x >> 5;
    if (lane == 0) red[wave] = v;
    __syncthreads();
    if (wave == 0) {
        v = (lane < 8) ? red[lane] : 0.0f;
#pragma unroll
        for (int off = 4; off; off >>= 1) v += __shfl_xor(v, off, 32);
    }
    return v;
}

__global__ __launch_bounds__(256) void zreduce(
    const float* __restrict__ x, float* __restrict__ part)
{
    const int total = BSZ * KTOT;
    float s = 0.0f;
    for (int i = blockIdx.x * blockDim.x + threadIdx.x; i < total;
         i += gridDim.x * blockDim.x)
        s += x[i];
    s = block_reduce(s);
    if (threadIdx.x == 0) part[blockIdx.x] = s;
}

__global__ void zfinal(const float* __restrict__ zs, const float* __restrict__ zt,
                       float* __restrict__ zval)
{
    if (threadIdx.x == 0) {
        double a = 0.0, b = 0.0;
        for (int i = 0; i < 1024; ++i) { a += zs[i]; b += zt[i]; }
        const double scale = (double)N_DATA_F / ((double)BSZ * (double)KTOT);
        zval[0] = (float)(a * scale);   // Z_s
        zval[1] = (float)(b * scale);   // Z_t
    }
}

__global__ __launch_bounds__(256) void loss_partial(
    const float* __restrict__ outx, const float* __restrict__ Zptr,
    float* __restrict__ lpart)
{
    const float Z   = *Zptr;
    const float mPn = 4096.0f / N_DATA_F;
    const int total = BSZ * KTOT;
    float s = 0.0f;
    for (int i = blockIdx.x * blockDim.x + threadIdx.x; i < total;
         i += gridDim.x * blockDim.x) {
        const float x = outx[i] / Z;
        const float denom = x + mPn + 1e-7f;
        const float num = ((i % KTOT) == 0) ? x : mPn;   // k==0 => log_D1 else log_D0
        s += __logf(num / denom);
    }
    s = block_reduce(s);
    if (threadIdx.x == 0) lpart[blockIdx.x] = s;
}

__global__ void finalize(const float* __restrict__ lp, float* __restrict__ out0)
{
    if (threadIdx.x == 0) {
        double s = 0.0;
        for (int i = 0; i < 2048; ++i) s += lp[i];
        out0[0] = (float)(-s / (double)BSZ);
    }
}

// ---------------------------------------------------------------------------
// Stream both memories to the output (1 GB): non-temporal b128 loads,
// scalar stores because new_m1 starts at d_out+1 (only 4B aligned).
// ---------------------------------------------------------------------------
__global__ __launch_bounds__(256) void copy_mem(
    const vf4* __restrict__ m1, const vf4* __restrict__ m2,
    float* __restrict__ o1, float* __restrict__ o2)
{
    const long long n4 = 500000LL * FEAT / 4;   // 16,000,000 float4 each
    const long long stride = (long long)gridDim.x * blockDim.x;
    for (long long i = (long long)blockIdx.x * blockDim.x + threadIdx.x;
         i < n4; i += stride) {
        const vf4 a = __builtin_nontemporal_load(&m1[i]);
        const vf4 b = __builtin_nontemporal_load(&m2[i]);
        const long long j = i << 2;
        __builtin_nontemporal_store(a.x, o1 + j);
        __builtin_nontemporal_store(a.y, o1 + j + 1);
        __builtin_nontemporal_store(a.z, o1 + j + 2);
        __builtin_nontemporal_store(a.w, o1 + j + 3);
        __builtin_nontemporal_store(b.x, o2 + j);
        __builtin_nontemporal_store(b.y, o2 + j + 1);
        __builtin_nontemporal_store(b.z, o2 + j + 2);
        __builtin_nontemporal_store(b.w, o2 + j + 3);
    }
}

// ---------------------------------------------------------------------------
// Update the 256 touched rows: new = 0.5*mem[idx] + 0.5*emb, L2-normalized.
// Last duplicate index wins (matches JAX .at[].set). One wave per row.
// ---------------------------------------------------------------------------
__global__ __launch_bounds__(256) void update_rows(
    const float* __restrict__ mem, const float* __restrict__ emb,
    const int* __restrict__ idx, float* __restrict__ outm)
{
    const int r = (blockIdx.x << 3) + (threadIdx.x >> 5);
    if (r >= BSZ) return;
    const int my = idx[r];
    for (int j = r + 1; j < BSZ; ++j)
        if (idx[j] == my) return;          // a later write supersedes this one
    const int lane = threadIdx.x & 31;
    const vf4 m = *(const vf4*)(mem + (size_t)my * FEAT + lane * 4);
    const vf4 e = *(const vf4*)(emb + (size_t)r  * FEAT + lane * 4);
    vf4 n;
    n.x = 0.5f * (m.x + e.x); n.y = 0.5f * (m.y + e.y);
    n.z = 0.5f * (m.z + e.z); n.w = 0.5f * (m.w + e.w);
    float s = n.x * n.x + n.y * n.y + n.z * n.z + n.w * n.w;
#pragma unroll
    for (int off = 16; off; off >>= 1) s += __shfl_xor(s, off, 32);
    const float inv = 1.0f / sqrtf(s);
    float* op = outm + (size_t)my * FEAT + lane * 4;   // only 4B-aligned base
    op[0] = n.x * inv; op[1] = n.y * inv; op[2] = n.z * inv; op[3] = n.w * inv;
}

// ---------------------------------------------------------------------------
extern "C" void kernel_launch(void* const* d_in, const int* in_sizes, int n_in,
                              void* d_out, int out_size, void* d_ws, size_t ws_size,
                              hipStream_t stream)
{
    (void)in_sizes; (void)n_in; (void)out_size; (void)ws_size;
    const float* f_s  = (const float*)d_in[0];
    const float* f_t  = (const float*)d_in[1];
    const int*   idx  = (const int*)  d_in[2];
    const int*   cidx = (const int*)  d_in[3];
    const float* W_s  = (const float*)d_in[4];
    const float* b_s  = (const float*)d_in[5];
    const float* W_t  = (const float*)d_in[6];
    const float* b_t  = (const float*)d_in[7];
    const float* m1   = (const float*)d_in[8];
    const float* m2   = (const float*)d_in[9];

    float* ws    = (float*)d_ws;
    float* es    = ws;                         // 256*128
    float* et    = ws + 32768;                 // 256*128
    float* out_s = ws + 65536;                 // 256*4097
    float* out_t = out_s + (size_t)BSZ * KTOT;
    float* zp_s  = out_t + (size_t)BSZ * KTOT; // 1024
    float* zp_t  = zp_s + 1024;                // 1024
    float* zval  = zp_t + 1024;                // [Z_s, Z_t] (+pad)
    float* lp    = zval + 16;                  // 2048 loss partials

    float* o   = (float*)d_out;
    float* om1 = o + 1;
    float* om2 = o + 1 + 500000LL * FEAT;

    // 1) projections (WMMA) + L2 norm
    proj_wmma<<<16, 256, 0, stream>>>(f_s, W_s, b_s, es, 1024);
    proj_wmma<<<16, 256, 0, stream>>>(f_t, W_t, b_t, et, 2048);
    l2norm_rows<<<64, 256, 0, stream>>>(ws, 512);  // es & et are contiguous

    // 2) NCE gather-dot-exp  (out_t: memory_v1 x et ; out_s: memory_v2 x es)
    nce_kernel<<<BSZ, 256, 0, stream>>>(m1, et, cidx, out_t);
    nce_kernel<<<BSZ, 256, 0, stream>>>(m2, es, cidx, out_s);

    // 3) Z + loss (deterministic tree reductions)
    zreduce<<<1024, 256, 0, stream>>>(out_s, zp_s);
    zreduce<<<1024, 256, 0, stream>>>(out_t, zp_t);
    zfinal<<<1, 32, 0, stream>>>(zp_s, zp_t, zval);
    loss_partial<<<1024, 256, 0, stream>>>(out_s, zval + 0, lp);
    loss_partial<<<1024, 256, 0, stream>>>(out_t, zval + 1, lp + 1024);
    finalize<<<1, 32, 0, stream>>>(lp, o);

    // 4) memory bank outputs: stream copy, then overwrite the 256 updated rows
    copy_mem<<<8192, 256, 0, stream>>>((const vf4*)m1, (const vf4*)m2, om1, om2);
    update_rows<<<32, 256, 0, stream>>>(m1, es, idx, om1);
    update_rows<<<32, 256, 0, stream>>>(m2, et, idx, om2);
}
